// MixerHead_5626407158009
// MI455X (gfx1250) — compile-verified, matrix operand order checked by reading, not run on hardware
//
#include <hip/hip_runtime.h>
#include <hip/hip_bf16.h>

// ---------------------------------------------------------------------------
// MixerHead on MI455X (gfx1250): three bf16-WMMA GEMMs, double-buffered LDS,
// async global->LDS copies (ASYNCcnt) for the bf16 operand tiles.
//   1) proj  = x @ Wp[h]^T + bp        -> projT (bf16, [b,h][e][l] transposed)
//   2) mixed = tril(Wc[h]) @ proj + bc -> hidden (bf16, [b][l][h*E+e])
//   3) out   = hidden @ Wo^T + bo      -> f32
// ---------------------------------------------------------------------------

typedef __bf16 bf16_t;
typedef __attribute__((ext_vector_type(4)))  __bf16 v4bf;
typedef __attribute__((ext_vector_type(8)))  __bf16 v8bf;
typedef __attribute__((ext_vector_type(16))) __bf16 v16bf;
typedef __attribute__((ext_vector_type(8)))  float  v8f;

constexpr int Bb = 4, Ls = 2048, Dd = 1024, Hh = 4, Ee = 256;

constexpr int BM = 128, BN = 128, BK = 64;
constexpr int BKP = BK + 16;   // padded LDS row: 160B => 16B-aligned rows, bank-shifted
constexpr int NT  = 256;       // 8 waves (wave32)

// round-to-nearest-even f32 -> bf16 (integer path: guaranteed to compile)
__device__ __forceinline__ bf16_t f2bf(float f) {
    unsigned u = __float_as_uint(f);
    u += 0x7FFFu + ((u >> 16) & 1u);
    unsigned short s = (unsigned short)(u >> 16);
    return __builtin_bit_cast(bf16_t, s);
}

__device__ __forceinline__ void wait_async0() {
#if __has_builtin(__builtin_amdgcn_s_wait_asynccnt)
    __builtin_amdgcn_s_wait_asynccnt(0);
#else
    asm volatile("s_wait_asynccnt 0x0" ::: "memory");
#endif
}

// A/B fragment per ISA 7.12.2 (16-bit, 16x32): lane&15 = row/col index,
// lane>>4 selects K halves {0..7,16..23} vs {8..15,24..31} => two 16B LDS loads.
__device__ __forceinline__ v16bf load_frag(const bf16_t* row, int half) {
    v8bf lo = *reinterpret_cast<const v8bf*>(row + 8 * half);
    v8bf hi = *reinterpret_cast<const v8bf*>(row + 16 + 8 * half);
    return __builtin_shufflevector(lo, hi,
        0,1,2,3,4,5,6,7,8,9,10,11,12,13,14,15);
}

// Stage a 128x64 f32 tile (row-major, leading dim ld) into LDS as bf16.
// MASK: causal tril mask, keep element (row0+r, k0+c) iff col <= row.
template<bool MASK>
__device__ __forceinline__ void stage_f32(bf16_t* s, const float* g, int ld,
                                          int row0, int k0) {
    const int t = threadIdx.x;
#pragma unroll
    for (int i = 0; i < (BM * BK / 4) / NT; ++i) {   // 8 iterations
        int idx = t + i * NT;
        int r   = idx >> 4;           // 16 float4 vectors per row
        int c   = (idx & 15) << 2;
        const float4 v =
            *reinterpret_cast<const float4*>(g + (size_t)(row0 + r) * ld + k0 + c);
        v4bf o;
        if (MASK) {
            int f = row0 + r, l = k0 + c;
            o[0] = f2bf((l + 0) <= f ? v.x : 0.0f);
            o[1] = f2bf((l + 1) <= f ? v.y : 0.0f);
            o[2] = f2bf((l + 2) <= f ? v.z : 0.0f);
            o[3] = f2bf((l + 3) <= f ? v.w : 0.0f);
        } else {
            o[0] = f2bf(v.x); o[1] = f2bf(v.y);
            o[2] = f2bf(v.z); o[3] = f2bf(v.w);
        }
        *reinterpret_cast<v4bf*>(s + r * BKP + c) = o;
    }
}

// Stage a 128x64 bf16 tile: pure copy -> async global->LDS DMA (no VGPR data,
// tracked on ASYNCcnt). Per lane: one 16B transfer per iteration.
__device__ __forceinline__ void stage_bf16_async(bf16_t* s, const bf16_t* g,
                                                 int ld, int row0, int k0) {
    const int t = threadIdx.x;
#pragma unroll
    for (int i = 0; i < (BM * BK / 8) / NT; ++i) {   // 4 iterations
        int idx = t + i * NT;
        int r   = idx >> 3;           // 8 x 16B per row
        int c   = (idx & 7) << 3;
        const bf16_t* gp  = g + (size_t)(row0 + r) * ld + k0 + c;
        // generic->LDS: low 32 bits of a __shared__ generic pointer are the
        // LDS byte offset (aperture in the high bits)
        unsigned lds_off = (unsigned)(size_t)(s + r * BKP + c);
        asm volatile("global_load_async_to_lds_b128 %0, %1, off"
                     :: "v"(lds_off), "v"(gp) : "memory");
    }
}

// One BK=64 step: 2 WMMA K-steps, wave computes 4x2 tiles of 16x16.
__device__ __forceinline__ void mma_step(const bf16_t* As, const bf16_t* Bs,
                                         v8f (&acc)[4][2],
                                         int wm, int wn, int half, int l16) {
#pragma unroll
    for (int kk = 0; kk < BK; kk += 32) {
        v16bf a[4], b[2];
#pragma unroll
        for (int i = 0; i < 4; ++i)
            a[i] = load_frag(As + (wm + i * 16 + l16) * BKP + kk, half);
#pragma unroll
        for (int j = 0; j < 2; ++j)
            b[j] = load_frag(Bs + (wn + j * 16 + l16) * BKP + kk, half);
#pragma unroll
        for (int i = 0; i < 4; ++i)
#pragma unroll
            for (int j = 0; j < 2; ++j)
                acc[i][j] = __builtin_amdgcn_wmma_f32_16x16x32_bf16(
                    false, a[i], false, b[j], (short)0, acc[i][j], false, false);
    }
}

// ---------------- Kernel 1: per-head input projection --------------------
// C[l][e] = sum_d x[b][l][d] * Wp[h][e][d]  (+bp) ; stored transposed [e][l]
__global__ __launch_bounds__(NT) void k_proj(const float* __restrict__ x,
                                             const float* __restrict__ Wp,
                                             const float* __restrict__ bp,
                                             bf16_t* __restrict__ projT) {
    __shared__ bf16_t As[2][BM * BKP];
    __shared__ bf16_t Bs[2][BN * BKP];
    const int m0 = blockIdx.x * BM;            // l
    const int n0 = blockIdx.y * BN;            // e
    const int h  = blockIdx.z % Hh;
    const int b  = blockIdx.z / Hh;
    const float* A  = x  + (size_t)b * Ls * Dd;      // [l][d]
    const float* Bm = Wp + (size_t)h * Ee * Dd;      // [e][d] : rows=n, cols=k

    const int wave = threadIdx.x >> 5, lane = threadIdx.x & 31;
    const int wm = (wave & 1) * 64, wn = (wave >> 1) * 32;
    const int half = lane >> 4, l16 = lane & 15;

    stage_f32<false>(As[0], A,  Dd, m0, 0);
    stage_f32<false>(Bs[0], Bm, Dd, n0, 0);
    __syncthreads();

    v8f acc[4][2] = {};
    int cur = 0;
    for (int k0 = 0; k0 < Dd; k0 += BK) {
        const int nxt = cur ^ 1;
        if (k0 + BK < Dd) {                    // prefetch next tile into other buffer
            stage_f32<false>(As[nxt], A,  Dd, m0, k0 + BK);
            stage_f32<false>(Bs[nxt], Bm, Dd, n0, k0 + BK);
        }
        mma_step(As[cur], Bs[cur], acc, wm, wn, half, l16);
        __syncthreads();
        cur = nxt;
    }

    bf16_t* outp = projT + (size_t)blockIdx.z * Ee * Ls;  // [e][l]
#pragma unroll
    for (int i = 0; i < 4; ++i)
#pragma unroll
        for (int j = 0; j < 2; ++j) {
            const int e = n0 + wn + j * 16 + l16;
            const int l = m0 + wm + i * 16 + 8 * half;   // 8 consecutive rows
            const float bias = bp[h * Ee + e];
            v8bf o;
#pragma unroll
            for (int r = 0; r < 8; ++r) o[r] = f2bf(acc[i][j][r] + bias);
            *reinterpret_cast<v8bf*>(outp + (size_t)e * Ls + l) = o;  // 16B store
        }
}

// ---------------- Kernel 2: causal token mixing --------------------------
// C[f][e] = sum_{l<=f} Wc[h][f][l] * proj[l][e]  (+bc) -> hidden[b][f][h*E+e]
__global__ __launch_bounds__(NT) void k_mix(const float* __restrict__ Wc,
                                            const float* __restrict__ bc,
                                            const bf16_t* __restrict__ projT,
                                            bf16_t* __restrict__ hidden) {
    __shared__ bf16_t As[2][BM * BKP];
    __shared__ bf16_t Bs[2][BN * BKP];
    const int m0 = blockIdx.x * BM;            // f
    const int n0 = blockIdx.y * BN;            // e
    const int h  = blockIdx.z % Hh;
    const int b  = blockIdx.z / Hh;
    const float*  A  = Wc + (size_t)h * Ls * Ls;             // [f][l]
    const bf16_t* Bm = projT + (size_t)blockIdx.z * Ee * Ls; // [e][l]

    const int wave = threadIdx.x >> 5, lane = threadIdx.x & 31;
    const int wm = (wave & 1) * 64, wn = (wave >> 1) * 32;
    const int half = lane >> 4, l16 = lane & 15;

    const int Klim = m0 + BM;                  // causal: skip tiles above diagonal
    stage_f32<true>(As[0], A, Ls, m0, 0);      // tril mask applied while staging
    stage_bf16_async(Bs[0], Bm, Ls, n0, 0);
    wait_async0();
    __syncthreads();

    v8f acc[4][2] = {};
    int cur = 0;
    for (int k0 = 0; k0 < Klim; k0 += BK) {
        const int nxt = cur ^ 1;
        if (k0 + BK < Klim) {
            stage_f32<true>(As[nxt], A, Ls, m0, k0 + BK);
            stage_bf16_async(Bs[nxt], Bm, Ls, n0, k0 + BK);
        }
        mma_step(As[cur], Bs[cur], acc, wm, wn, half, l16);
        wait_async0();
        __syncthreads();
        cur = nxt;
    }

    bf16_t* outp = hidden + (size_t)b * Ls * Dd + h * Ee;  // concat heads
#pragma unroll
    for (int i = 0; i < 4; ++i)
#pragma unroll
        for (int j = 0; j < 2; ++j) {
            const int e  = n0 + wn + j * 16 + l16;
            const int f0 = m0 + wm + i * 16 + 8 * half;
#pragma unroll
            for (int r = 0; r < 8; ++r) {
                const int f = f0 + r;
                outp[(size_t)f * Dd + e] = f2bf(acc[i][j][r] + bc[h * Ls + f]);
            }
        }
}

// ---------------- Kernel 3: output projection ----------------------------
// out[m][n] = sum_k hidden[m][k] * Wo[n][k] + bo[n]   (m over B*L)
__global__ __launch_bounds__(NT) void k_out(const bf16_t* __restrict__ hidden,
                                            const float* __restrict__ Wo,
                                            const float* __restrict__ bo,
                                            float* __restrict__ out) {
    __shared__ bf16_t As[2][BM * BKP];
    __shared__ bf16_t Bs[2][BN * BKP];
    const int m0 = blockIdx.x * BM;
    const int n0 = blockIdx.y * BN;

    const int wave = threadIdx.x >> 5, lane = threadIdx.x & 31;
    const int wm = (wave & 1) * 64, wn = (wave >> 1) * 32;
    const int half = lane >> 4, l16 = lane & 15;

    stage_bf16_async(As[0], hidden, Dd, m0, 0);
    stage_f32<false>(Bs[0], Wo, Dd, n0, 0);    // Wo[n][k], contiguous in k
    wait_async0();
    __syncthreads();

    v8f acc[4][2] = {};
    int cur = 0;
    for (int k0 = 0; k0 < Dd; k0 += BK) {
        const int nxt = cur ^ 1;
        if (k0 + BK < Dd) {
            stage_bf16_async(As[nxt], hidden, Dd, m0, k0 + BK);
            stage_f32<false>(Bs[nxt], Wo, Dd, n0, k0 + BK);
        }
        mma_step(As[cur], Bs[cur], acc, wm, wn, half, l16);
        wait_async0();
        __syncthreads();
        cur = nxt;
    }

#pragma unroll
    for (int i = 0; i < 4; ++i)
#pragma unroll
        for (int j = 0; j < 2; ++j) {
            const int n  = n0 + wn + j * 16 + l16;
            const int mb = m0 + wm + i * 16 + 8 * half;
            const float bias = bo[n];
#pragma unroll
            for (int r = 0; r < 8; ++r)
                out[(size_t)(mb + r) * Dd + n] = acc[i][j][r] + bias;
        }
}

// ---------------------------------------------------------------------------
extern "C" void kernel_launch(void* const* d_in, const int* in_sizes, int n_in,
                              void* d_out, int out_size, void* d_ws, size_t ws_size,
                              hipStream_t stream) {
    const float* x  = (const float*)d_in[0];
    const float* Wp = (const float*)d_in[1];
    const float* bp = (const float*)d_in[2];
    const float* Wc = (const float*)d_in[3];
    const float* bc = (const float*)d_in[4];
    const float* Wo = (const float*)d_in[5];
    const float* bo = (const float*)d_in[6];
    float* out = (float*)d_out;

    bf16_t* projT  = (bf16_t*)d_ws;                         // 16 MB
    bf16_t* hidden = projT + (size_t)Bb * Hh * Ee * Ls;     // 16 MB

    dim3 blk(NT);
    k_proj<<<dim3(Ls / BM, Ee / BN, Bb * Hh), blk, 0, stream>>>(x, Wp, bp, projT);
    k_mix <<<dim3(Ls / BM, Ee / BN, Bb * Hh), blk, 0, stream>>>(Wc, bc, projT, hidden);
    k_out <<<dim3((Bb * Ls) / BM, Dd / BN, 1), blk, 0, stream>>>(hidden, Wo, bo, out);
}
